// CrossAttentionBlock_75986561401090
// MI455X (gfx1250) — matrix operand order repaired and checked by reading, not compiled, not run on hardware
//
#include <hip/hip_runtime.h>

// ---------------- types ----------------
typedef __attribute__((ext_vector_type(16))) _Float16 v16h;
typedef __attribute__((ext_vector_type(8)))  _Float16 v8h;
typedef __attribute__((ext_vector_type(8)))  float    v8f;
typedef __attribute__((ext_vector_type(2)))  float    f32x2;
typedef __attribute__((ext_vector_type(4)))  unsigned int u32x4;
typedef __attribute__((ext_vector_type(8)))  int          i32x8;
typedef __attribute__((ext_vector_type(4)))  int          i32x4;

union HV { v16h v; v8h h[2]; };

// ---------------- problem constants ----------------
#define B_    2
#define NTOK  2048
#define DIM   1024
#define H_    16
#define HD_   64
#define SCALE_ 0.125f   // 64^-0.5

// =====================================================================
// TDM: async 2D tile load global->LDS (Tensor Data Mover).
// Tile: 32 rows x 64 f16 (128B/row), row stride DIM elements in memory,
// LDS rows padded to 144B (pad 4 dwords every 32 dwords) for bank spread.
// D# built per CDNA5 ISA. 2D tensor: groups 2,3 zero.
// This toolchain exposes the 6-arg builtin (extra i32x8 before cpol).
// =====================================================================
__device__ __forceinline__ void tdm_load_k_tile(const _Float16* gptr, void* ldsptr) {
    unsigned long long ga = (unsigned long long)(uintptr_t)gptr;
    unsigned int       la = (unsigned int)(uintptr_t)ldsptr;   // low 32b = LDS offset
    u32x4 g0;
    g0[0] = 1u;                                        // count=1, user descriptor
    g0[1] = la;                                        // lds_addr
    g0[2] = (unsigned int)ga;                          // global_addr[31:0]
    g0[3] = (unsigned int)((ga >> 32) & 0x01FFFFFFu)   // global_addr[56:32]
          | (2u << 30);                                // type = 2 ("image")
    i32x8 g1;
    g1[0] = (1 << 16)                                  // data_size = 2 bytes
          | (1 << 20)                                  // pad_enable
          | (4 << 22)                                  // pad_interval: 32 dwords (128B)
          | (3 << 25);                                 // pad_amount: 4 dwords (16B)
    g1[1] = (HD_ & 0xFFFF) << 16;                      // tensor_dim0 = 64 (lo half)
    g1[2] = (32 << 16);                                // dim0 hi=0 | tensor_dim1 = 32
    g1[3] = (HD_ << 16);                               // dim1 hi=0 | tile_dim0 = 64
    g1[4] = 32;                                        // tile_dim1 = 32, tile_dim2 = 0
    g1[5] = DIM;                                       // tensor_dim0_stride = 1024
    g1[6] = 0;                                         // stride hi | dim1_stride lo
    g1[7] = 0;
    i32x4 z4 = {0, 0, 0, 0};
    i32x8 z8 = {0, 0, 0, 0, 0, 0, 0, 0};
    __builtin_amdgcn_tensor_load_to_lds(g0, g1, z4, z4, z8, 0);
}

// =====================================================================
// Kernel 0: convert qkv_w (f32 [3D, D]) -> f16 in workspace
// =====================================================================
__global__ __launch_bounds__(256) void cvt_w(const float* __restrict__ w,
                                             _Float16* __restrict__ wh) {
    int i = (blockIdx.x * 256 + threadIdx.x) * 4;
    float4 v = *(const float4*)(w + i);
    wh[i + 0] = (_Float16)v.x;
    wh[i + 1] = (_Float16)v.y;
    wh[i + 2] = (_Float16)v.z;
    wh[i + 3] = (_Float16)v.w;
}

// =====================================================================
// Kernel 1: fused QKV GEMM.  out[M=B*N, E=3D] = x @ W^T + b  (f16 out)
// Block: 256 threads = 8 waves; block tile = 16 (M) x 512 (E).
// Wave tile = 16 x 64: one A fragment feeds 4 WMMAs per K-step.
// qkvh layout: [proj(3)][b][tok][d]  (f16)
// =====================================================================
__global__ __launch_bounds__(256) void qkv_gemm(const float* __restrict__ x1,
                                                const float* __restrict__ x2,
                                                const _Float16* __restrict__ wh,
                                                const float* __restrict__ bias,
                                                _Float16* __restrict__ qkvh) {
    __shared__ alignas(16) _Float16 At[16][40];   // 16 rows x 32 k, pad->40

    const int t      = threadIdx.x;
    const int wv     = t >> 5;
    const int lane   = t & 31;
    const int ln15   = lane & 15;
    const int hiHalf = lane >> 4;              // 0 or 1
    const int kbA    = hiHalf ? 8  : 0;        // unused: A fragment also loads at kbB
    const int kbB    = hiHalf ? 16 : 0;        // B-matrix k base per lane half
    const int rowOff = hiHalf ? 8  : 0;        // C/D row base per lane half

    const int mt  = blockIdx.x;                // 0..255  (row tile)
    const int nt0 = blockIdx.y * 32 + wv * 4;  // first 16-col tile of this wave

    const int b    = (mt * 16) / NTOK;
    const int tok0 = (mt * 16) % NTOK;
    // proj 0 (Q, cols < D) reads x1; proj 1,2 (K,V) read x2. 512 | 1024.
    const float* __restrict__ src = (blockIdx.y < 2) ? x1 : x2;

    const _Float16* __restrict__ wrow[4];
#pragma unroll
    for (int s = 0; s < 4; ++s)
        wrow[s] = wh + (size_t)((nt0 + s) * 16 + ln15) * DIM;

    v8f acc[4] = {};

    for (int ks = 0; ks < DIM; ks += 32) {
        __syncthreads();                       // protect At from prior reads
        {   // stage A tile: 16x32 f32 -> f16 LDS (2 elems/thread, coalesced)
            int e = t * 2;
            int r = e >> 5, c = e & 31;
            f32x2 xv = *(const f32x2*)(src + ((size_t)b * NTOK + tok0 + r) * DIM + ks + c);
            At[r][c]     = (_Float16)xv.x;
            At[r][c + 1] = (_Float16)xv.y;
        }
        __syncthreads();

        HV a;
        a.h[0] = *(const v8h*)&At[ln15][kbB];            // k = kbB..kbB+7
        a.h[1] = *(const v8h*)&At[ln15][kbB + 8];        // k = kbB+8..+15
#pragma unroll
        for (int s = 0; s < 4; ++s) {
            HV bf;
            bf.h[0] = *(const v8h*)(wrow[s] + ks + kbB);
            bf.h[1] = *(const v8h*)(wrow[s] + ks + kbB + 8);
            acc[s] = __builtin_amdgcn_wmma_f32_16x16x32_f16(false, a.v, false, bf.v,
                                                            (short)0, acc[s], false, false);
        }
    }

    // epilogue: + bias, cast f16, store into [proj][b][tok][d]
#pragma unroll
    for (int s = 0; s < 4; ++s) {
        const int   ecol = (nt0 + s) * 16 + ln15;
        const float bv   = bias[ecol];
        const int   proj = ecol >> 10;
        const int   d    = ecol & (DIM - 1);
        _Float16* __restrict__ dst = qkvh + (size_t)proj * (B_ * NTOK * DIM);
#pragma unroll
        for (int r = 0; r < 8; ++r) {
            int tok = tok0 + r + rowOff;
            dst[((size_t)b * NTOK + tok) * DIM + d] = (_Float16)(acc[s][r] + bv);
        }
    }
}

// =====================================================================
// Kernel 2: flash attention per (b, h).  Block = 128 threads = 4 waves.
// Each wave owns 16 query rows; block covers 64 rows. Key loop: 32 keys/step.
// K tile staged by the Tensor Data Mover (async); V staged transposed by ALUs.
// =====================================================================
__global__ __launch_bounds__(128) void attn(const _Float16* __restrict__ qkvh,
                                            float* __restrict__ out) {
    __shared__ alignas(16) _Float16 Kt[32][72];       // [tok_local][qk], 144B rows (TDM-padded)
    __shared__ alignas(16) _Float16 Vt[HD_][40];      // transposed: [hd][tok_local]
    __shared__ alignas(16) _Float16 Pt[4][16][40];    // per-wave P transpose buffer

    const int t      = threadIdx.x;
    const int wv     = t >> 5;
    const int lane   = t & 31;
    const int ln15   = lane & 15;
    const int hiHalf = lane >> 4;
    const int kbA    = hiHalf ? 8  : 0;
    const int kbB    = hiHalf ? 16 : 0;
    const int rowOff = hiHalf ? 8  : 0;

    const int bh = blockIdx.y;
    const int b  = bh / H_;
    const int hh = bh % H_;
    const int q0 = blockIdx.x * 64 + wv * 16;

    const _Float16* __restrict__ Qh = qkvh;
    const _Float16* __restrict__ Kh = qkvh + (size_t)1 * B_ * NTOK * DIM;
    const _Float16* __restrict__ Vh = qkvh + (size_t)2 * B_ * NTOK * DIM;

    // ---- load the wave's Q fragments once (16 rows x 64 qk) ----
    const _Float16* qrow = Qh + ((size_t)b * NTOK + q0 + ln15) * DIM + hh * HD_;
    HV qf[2];
#pragma unroll
    for (int f = 0; f < 2; ++f) {
        qf[f].h[0] = *(const v8h*)(qrow + f * 32 + kbB);
        qf[f].h[1] = *(const v8h*)(qrow + f * 32 + kbB + 8);
    }

    float mrow[8], lrow[8];
    v8f acc[4] = {};
#pragma unroll
    for (int r = 0; r < 8; ++r) { mrow[r] = -3.0e38f; lrow[r] = 0.0f; }

    const int stage_r = t >> 2;          // 0..31 token row
    const int stage_c = (t & 3) * 16;    // 0,16,32,48 hd column base

    for (int jt = 0; jt < NTOK / 32; ++jt) {
        const int j = jt * 32;

        // ---- async K tile via TDM (one wave issues; EXEC ignored by TDM) ----
        if (wv == 0)
            tdm_load_k_tile(Kh + ((size_t)b * NTOK + j) * DIM + hh * HD_, &Kt[0][0]);

        // ---- stage V tile transposed into LDS (all threads) ----
        {
            const _Float16* vrow = Vh + ((size_t)b * NTOK + j + stage_r) * DIM + hh * HD_ + stage_c;
            v8h v0 = *(const v8h*)(vrow);
            v8h v1 = *(const v8h*)(vrow + 8);
#pragma unroll
            for (int e = 0; e < 8; ++e) {
                Vt[stage_c + e][stage_r]     = v0[e];
                Vt[stage_c + 8 + e][stage_r] = v1[e];
            }
        }
        if (jt + 1 < NTOK / 32)   // prefetch next V tile -> global_prefetch_b8
            __builtin_prefetch(Vh + ((size_t)b * NTOK + j + 32 + stage_r) * DIM + hh * HD_ + stage_c, 0, 1);

        __builtin_amdgcn_s_wait_tensorcnt(0);   // TDM done (no-op for waves 1-3)
        __syncthreads();

        // ---- S = (Q K^T) * scale : two 16-col halves, K-dim 64 = 2 wmma each
        v8f s0 = {}, s1 = {};
#pragma unroll
        for (int f = 0; f < 2; ++f) {
            HV kf0, kf1;
            kf0.h[0] = *(const v8h*)&Kt[ln15][f * 32 + kbB];
            kf0.h[1] = *(const v8h*)&Kt[ln15][f * 32 + kbB + 8];
            kf1.h[0] = *(const v8h*)&Kt[16 + ln15][f * 32 + kbB];
            kf1.h[1] = *(const v8h*)&Kt[16 + ln15][f * 32 + kbB + 8];
            s0 = __builtin_amdgcn_wmma_f32_16x16x32_f16(false, qf[f].v, false, kf0.v,
                                                        (short)0, s0, false, false);
            s1 = __builtin_amdgcn_wmma_f32_16x16x32_f16(false, qf[f].v, false, kf1.v,
                                                        (short)0, s1, false, false);
        }

        // ---- online softmax (rows are wave-private; reduce over 16 lanes) ----
        float p0[8], p1[8];
#pragma unroll
        for (int r = 0; r < 8; ++r) {
            float a0 = s0[r] * SCALE_;
            float a1 = s1[r] * SCALE_;
            float mx = fmaxf(a0, a1);
            mx = fmaxf(mx, __shfl_xor(mx, 1, 16));
            mx = fmaxf(mx, __shfl_xor(mx, 2, 16));
            mx = fmaxf(mx, __shfl_xor(mx, 4, 16));
            mx = fmaxf(mx, __shfl_xor(mx, 8, 16));
            float mnew  = fmaxf(mrow[r], mx);
            float alpha = __expf(mrow[r] - mnew);
            float e0 = __expf(a0 - mnew);
            float e1 = __expf(a1 - mnew);
            float rs = e0 + e1;
            rs += __shfl_xor(rs, 1, 16);
            rs += __shfl_xor(rs, 2, 16);
            rs += __shfl_xor(rs, 4, 16);
            rs += __shfl_xor(rs, 8, 16);
            lrow[r] = lrow[r] * alpha + rs;
            mrow[r] = mnew;
            p0[r] = e0;
            p1[r] = e1;
#pragma unroll
            for (int s = 0; s < 4; ++s) acc[s][r] *= alpha;
        }

        // ---- C-layout -> A-layout transpose of P through per-wave LDS ----
#pragma unroll
        for (int r = 0; r < 8; ++r) {
            Pt[wv][r + rowOff][ln15]      = (_Float16)p0[r];
            Pt[wv][r + rowOff][16 + ln15] = (_Float16)p1[r];
        }
        asm volatile("s_wait_dscnt 0" ::: "memory");   // wave-local LDS RAW fence

        HV pf;
        pf.h[0] = *(const v8h*)&Pt[wv][ln15][kbB];
        pf.h[1] = *(const v8h*)&Pt[wv][ln15][kbB + 8];

        // ---- acc += P @ V  (4 hd segments of 16) ----
#pragma unroll
        for (int s = 0; s < 4; ++s) {
            HV vf;
            vf.h[0] = *(const v8h*)&Vt[s * 16 + ln15][kbB];
            vf.h[1] = *(const v8h*)&Vt[s * 16 + ln15][kbB + 8];
            acc[s] = __builtin_amdgcn_wmma_f32_16x16x32_f16(false, pf.v, false, vf.v,
                                                            (short)0, acc[s], false, false);
        }
        __syncthreads();    // K/V tiles reusable next iteration
    }

    // ---- epilogue: out[b, tok, h*HD + hd] = acc / l ----
#pragma unroll
    for (int r = 0; r < 8; ++r) {
        float inv = 1.0f / lrow[r];
        int   tok = q0 + r + rowOff;
        float* orow = out + ((size_t)b * NTOK + tok) * DIM + hh * HD_;
#pragma unroll
        for (int s = 0; s < 4; ++s)
            orow[s * 16 + ln15] = acc[s][r] * inv;
    }
}

// =====================================================================
// launch
// =====================================================================
extern "C" void kernel_launch(void* const* d_in, const int* in_sizes, int n_in,
                              void* d_out, int out_size, void* d_ws, size_t ws_size,
                              hipStream_t stream) {
    const float* x1    = (const float*)d_in[0];
    const float* x2    = (const float*)d_in[1];
    const float* qkv_w = (const float*)d_in[2];
    const float* qkv_b = (const float*)d_in[3];
    float*       out   = (float*)d_out;

    // workspace: Wh (f16, 3D*D = 6.3MB) then QKVh (f16, 3*B*N*D = 25.2MB)
    _Float16* wh   = (_Float16*)d_ws;
    _Float16* qkvh = wh + (size_t)3 * DIM * DIM;

    // 0) weights f32 -> f16
    cvt_w<<<(3 * DIM * DIM) / (256 * 4), 256, 0, stream>>>(qkv_w, wh);

    // 1) fused QKV projection: M tiles = B*N/16 = 256, E groups = 3072/512 = 6
    dim3 g1(256, 6);
    qkv_gemm<<<g1, 256, 0, stream>>>(x1, x2, wh, qkv_b, qkvh);

    // 2) attention: 32 q-blocks of 64 rows x 32 (b,h) pairs
    dim3 g2(NTOK / 64, B_ * H_);
    attn<<<g2, 128, 0, stream>>>(qkvh, out);
}
